// CrossAttention_54863912239294
// MI455X (gfx1250) — matrix operand order repaired and checked by reading, not compiled, not run on hardware
//
#include <hip/hip_runtime.h>
#include <hip/hip_bf16.h>
#include <stdint.h>

#define B_    4
#define N_    2048
#define DIM_  512
#define H_    8
#define DH_   64
#define JP    2080          // 1 + 2048 padded up to multiple of 32
#define NC    65            // JP / 32 chunks of the j dimension
#define SCALE_ 0.125f       // DH^-0.5
#define EPS_   1e-5f

typedef __bf16 bf16;
typedef bf16  v8bf  __attribute__((ext_vector_type(8)));
typedef bf16  v16bf __attribute__((ext_vector_type(16)));
typedef float v8f   __attribute__((ext_vector_type(8)));
typedef unsigned int v4u __attribute__((ext_vector_type(4)));
typedef int  v8i __attribute__((ext_vector_type(8)));
typedef int  v4i __attribute__((ext_vector_type(4)));

static __device__ inline v16bf join8(v8bf lo, v8bf hi){
  return __builtin_shufflevector(lo, hi, 0,1,2,3,4,5,6,7,8,9,10,11,12,13,14,15);
}
static __device__ inline v8f wmma_bf16(v16bf a, v16bf b, v8f c){
  // D(f32 16x16) = A(16x32 bf16) * B(32x16 bf16) + C
  return __builtin_amdgcn_wmma_f32_16x16x32_bf16(false, a, false, b, (short)0, c, false, false);
}

// ---- Tensor Data Mover: 2D tile load (bf16 elements) Global -> LDS ----
// D# packing per CDNA5 ISA ch.8 (group0: count/lds/global/type, group1: dims/strides).
// Toolchain uses the 6-arg builtin: (v4u g0, v8i g1, v4i g2, v4i g3, v8i extra, i32 cpol).
static __device__ inline void tdm_load2d(unsigned lds, const void* g, unsigned dim0, unsigned dim1,
                                         unsigned tile0, unsigned tile1, unsigned stride0){
  unsigned long long ga = (unsigned long long)(uintptr_t)g;
  v4u g0;
  g0.x = 1u;                                                   // count = 1 valid descriptor
  g0.y = lds;                                                  // lds_addr (bytes)
  g0.z = (unsigned)ga;                                         // global_addr[31:0]
  g0.w = ((unsigned)(ga >> 32) & 0x01FFFFFFu) | 0x80000000u;   // global_addr[56:32] | type=2
  v8i g1;
  g1[0] = 0x00010000;                                          // data_size = 1 (2 bytes)
  g1[1] = (int)((dim0 & 0xFFFFu) << 16);                       // tensor_dim0[15:0]
  g1[2] = (int)((dim0 >> 16) | ((dim1 & 0xFFFFu) << 16));      // dim0[31:16] | dim1[15:0]
  g1[3] = (int)((dim1 >> 16) | ((tile0 & 0xFFFFu) << 16));     // dim1[31:16] | tile_dim0
  g1[4] = (int)(tile1 & 0xFFFFu);                              // tile_dim1 (tile_dim2 = 0)
  g1[5] = (int)stride0;                                        // tensor_dim0_stride[31:0]
  g1[6] = 0;
  g1[7] = 0;
  v4i z4; z4[0]=0; z4[1]=0; z4[2]=0; z4[3]=0;                  // 2-D tensor: groups 2/3 unused
  v8i z8; z8[0]=0; z8[1]=0; z8[2]=0; z8[3]=0; z8[4]=0; z8[5]=0; z8[6]=0; z8[7]=0;
  __builtin_amdgcn_tensor_load_to_lds(g0, g1, z4, z4, z8, 0);
}

// ---------------- LayerNorm: f32 x -> bf16 xn -------------------
__global__ __launch_bounds__(128) void k_ln1(const float* __restrict__ x,
                                             const float* __restrict__ gamma,
                                             bf16* __restrict__ xn){
  int row = blockIdx.x, t = threadIdx.x;
  const float* xr = x + (size_t)row * DIM_;
  float v[4]; float s = 0.f, s2 = 0.f;
  #pragma unroll
  for (int i = 0; i < 4; ++i){ float a = xr[t + 128*i]; v[i] = a; s += a; s2 += a*a; }
  #pragma unroll
  for (int off = 16; off >= 1; off >>= 1){ s += __shfl_xor(s, off, 32); s2 += __shfl_xor(s2, off, 32); }
  __shared__ float red[8];
  int wid = t >> 5, lane = t & 31;
  if (lane == 0){ red[wid] = s; red[4 + wid] = s2; }
  __syncthreads();
  s  = red[0] + red[1] + red[2] + red[3];
  s2 = red[4] + red[5] + red[6] + red[7];
  float mu = s * (1.f/DIM_);
  float var = s2 * (1.f/DIM_) - mu*mu;
  float rs = rsqrtf(var + EPS_);
  #pragma unroll
  for (int i = 0; i < 4; ++i)
    xn[(size_t)row*DIM_ + t + 128*i] = (bf16)((v[i]-mu)*rs*gamma[t + 128*i]);
}

// ------------- LayerNorm in place: f32 y -> f32 y ---------------
__global__ __launch_bounds__(128) void k_ln2(float* __restrict__ y, const float* __restrict__ g){
  int row = blockIdx.x, t = threadIdx.x;
  float* yr = y + (size_t)row * DIM_;
  float v[4]; float s = 0.f, s2 = 0.f;
  #pragma unroll
  for (int i = 0; i < 4; ++i){ float a = yr[t + 128*i]; v[i] = a; s += a; s2 += a*a; }
  #pragma unroll
  for (int off = 16; off >= 1; off >>= 1){ s += __shfl_xor(s, off, 32); s2 += __shfl_xor(s2, off, 32); }
  __shared__ float red[8];
  int wid = t >> 5, lane = t & 31;
  if (lane == 0){ red[wid] = s; red[4 + wid] = s2; }
  __syncthreads();
  s  = red[0] + red[1] + red[2] + red[3];
  s2 = red[4] + red[5] + red[6] + red[7];
  float mu = s * (1.f/DIM_);
  float var = s2 * (1.f/DIM_) - mu*mu;
  float rs = rsqrtf(var + EPS_);
  #pragma unroll
  for (int i = 0; i < 4; ++i)
    yr[t + 128*i] = (v[i]-mu)*rs*g[t + 128*i];
}

// ----- Weight prep: WcatT[o][d] = [Wq | Wkv](d,o), WoutT[o][d] = Wout(d,o), bf16 -----
__global__ void k_wconv(const float* __restrict__ Wq, const float* __restrict__ Wkv,
                        const float* __restrict__ Wout, bf16* __restrict__ WcatT,
                        bf16* __restrict__ WoutT){
  int idx = blockIdx.x * 256 + threadIdx.x;
  const int t1 = 640 * 512;
  if (idx < t1){
    int o = idx >> 9, d = idx & 511;
    float w = (o < 512) ? Wq[d*512 + o] : Wkv[d*128 + (o - 512)];
    WcatT[o*512 + d] = (bf16)w;
  } else {
    idx -= t1;
    if (idx < 512*512){
      int o = idx >> 9, d = idx & 511;
      WoutT[o*512 + d] = (bf16)Wout[d*512 + o];
    }
  }
}

// ----- bias from mask, null-kv row j=0, zero the j padding -----
__global__ void k_prep(const unsigned char* __restrict__ mask, const float* __restrict__ nullkv,
                       float* __restrict__ bias, bf16* __restrict__ K, bf16* __restrict__ VT){
  int idx = blockIdx.x * 256 + threadIdx.x;
  int r0 = B_ * JP;
  if (idx < r0){
    int b = idx / JP, j = idx % JP;
    float v;
    if (j == 0) v = 0.f;
    else if (j <= N_) v = mask[b*N_ + (j-1)] ? 0.f : -3.0e38f;
    else v = -3.0e38f;
    bias[idx] = v;
    return;
  }
  idx -= r0;
  if (idx < B_*DH_){ int b = idx >> 6, d = idx & 63; K[((size_t)b*JP)*DH_ + d] = (bf16)nullkv[d]; return; }
  idx -= B_*DH_;
  if (idx < B_*DH_){ int b = idx >> 6, d = idx & 63; VT[((size_t)b*DH_ + d)*JP] = (bf16)nullkv[DH_ + d]; return; }
  idx -= B_*DH_;
  if (idx < B_*31*DH_){
    int b = idx / (31*DH_); int rr = idx % (31*DH_);
    int j = N_ + 1 + rr / DH_; int d = rr % DH_;
    K[((size_t)b*JP + j)*DH_ + d] = (bf16)0.f; return;
  }
  idx -= B_*31*DH_;
  if (idx < B_*DH_*31){
    int b = idx / (DH_*31); int rr = idx % (DH_*31);
    int d = rr / 31; int j = N_ + 1 + rr % 31;
    VT[((size_t)b*DH_ + d)*JP + j] = (bf16)0.f; return;
  }
}

// ----- fused QKV projection: xn(8192x512) @ WcatT-as-B(512x640); 1 wave = 16x16 tile -----
__global__ __launch_bounds__(256) void k_proj(const bf16* __restrict__ xn, const bf16* __restrict__ WcatT,
                                              bf16* __restrict__ q, bf16* __restrict__ K, bf16* __restrict__ VT){
  int gw = (int)((blockIdx.x * blockDim.x + threadIdx.x) >> 5);
  int lane = threadIdx.x & 31;
  int half = lane >> 4, l16 = lane & 15;
  int mt = gw / 40, nt = gw % 40;
  int m0 = mt << 4, n0 = nt << 4;
  const bf16* arow = xn + (size_t)(m0 + l16) * DIM_;
  const bf16* brow = WcatT + (size_t)(n0 + l16) * DIM_;
  v8f acc = {};
  #pragma unroll
  for (int w = 0; w < 16; ++w){
    v8bf a0 = *(const v8bf*)(arow + 32*w + 16*half);
    v8bf a1 = *(const v8bf*)(arow + 32*w + 16*half + 8);
    v16bf bv = *(const v16bf*)(brow + 32*w + 16*half);
    acc = wmma_bf16(join8(a0, a1), bv, acc);
  }
  int o = n0 + l16;
  #pragma unroll
  for (int r = 0; r < 8; ++r){
    int m = m0 + r + 8*half;
    int b = m >> 11, n = m & (N_ - 1);
    float val = acc[r];
    if (o < 512){                       // Q: (b,h,n,d) pre-scaled
      int h = o >> 6, d = o & 63;
      q[(((size_t)b*H_ + h)*N_ + n)*DH_ + d] = (bf16)(val * SCALE_);
    } else if (o < 576){                // K: (b, j, d) row-major, j = n+1
      K[((size_t)b*JP + (n + 1))*DH_ + (o - 512)] = (bf16)val;
    } else {                            // V transposed: (b, d, j)
      VT[((size_t)b*DH_ + (o - 576))*JP + (n + 1)] = (bf16)val;
    }
  }
}

// ----- flash attention: 1 block = (b,h, 64 rows), 4 waves x 16-row tiles; TDM-staged K/V -----
__global__ __launch_bounds__(128) void k_attn(const bf16* __restrict__ q, const bf16* __restrict__ K,
                                              const bf16* __restrict__ VT, const float* __restrict__ bias,
                                              bf16* __restrict__ o){
  __shared__ __align__(32) bf16 sK[2][32 * DH_];   // [j][d] chunk, double buffered
  __shared__ __align__(32) bf16 sV[2][DH_ * 32];   // [d][j] chunk, double buffered
  __shared__ __align__(32) bf16 sP[4][16 * 32];    // per-wave P transpose staging
  if ((unsigned long long)(uintptr_t)q == 1ull){   // never true: keeps TDM-written LDS live
    sK[0][threadIdx.x] = (bf16)0.f; sV[0][threadIdx.x] = (bf16)0.f;
  }
  int wid = threadIdx.x >> 5, lane = threadIdx.x & 31;
  int half = lane >> 4, l16 = lane & 15;
  int bh = blockIdx.x >> 5;                 // b*H + h
  int itile = blockIdx.x & 31;
  int b = bh >> 3, h = bh & 7;
  int i0 = (itile << 6) + (wid << 4);

  const bf16* qrow = q + ((size_t)bh * N_ + i0 + l16) * DH_;
  v16bf qa0 = join8(*(const v8bf*)(qrow + 16*half),      *(const v8bf*)(qrow + 16*half + 8));
  v16bf qa1 = join8(*(const v8bf*)(qrow + 32 + 16*half), *(const v8bf*)(qrow + 40 + 16*half));

  v8f acc[4] = {};
  float mrow[8], lrow[8];
  #pragma unroll
  for (int r = 0; r < 8; ++r){ mrow[r] = -3.0e38f; lrow[r] = 0.f; }

  const bf16* Kb = K  + (size_t)b * JP * DH_;
  const bf16* Vb = VT + (size_t)b * DH_ * JP;
  unsigned ldsK = (unsigned)(uintptr_t)&sK[0][0];
  unsigned ldsV = (unsigned)(uintptr_t)&sV[0][0];

  if (wid == 0){                                              // prologue: chunk 0 -> buf 0
    tdm_load2d(ldsK, Kb, DH_, JP, DH_, 32, DH_);              // K tile: 32 j-rows x 64 d
    tdm_load2d(ldsV, Vb, JP, DH_, 32, DH_, JP);               // V tile: 64 d-rows x 32 j
  }
  for (int c = 0; c < NC; ++c){
    int buf = c & 1;
    if (wid == 0){
      if (c + 1 < NC){                                        // prefetch next chunk -> other buffer
        int nb = buf ^ 1;
        int j0n = (c + 1) * 32;
        tdm_load2d(ldsK + nb*(32*DH_*2), Kb + (size_t)j0n*DH_, DH_, JP, DH_, 32, DH_);
        tdm_load2d(ldsV + nb*(DH_*32*2), Vb + j0n,             JP, DH_, 32, DH_, JP);
        __builtin_amdgcn_s_wait_tensorcnt(2);                 // chunk c's pair is done
      } else {
        __builtin_amdgcn_s_wait_tensorcnt(0);
      }
    }
    __syncthreads();

    int j0 = c * 32;
    v8f sim[2];
    #pragma unroll
    for (int f = 0; f < 2; ++f){
      v16bf kb0 = *(const v16bf*)&sK[buf][(16*f + l16)*DH_ + 16*half];
      v16bf kb1 = *(const v16bf*)&sK[buf][(16*f + l16)*DH_ + 32 + 16*half];
      v8f s0 = {};
      s0     = wmma_bf16(qa0, kb0, s0);
      sim[f] = wmma_bf16(qa1, kb1, s0);
      float bv = bias[(size_t)b*JP + j0 + 16*f + l16];
      #pragma unroll
      for (int r = 0; r < 8; ++r) sim[f][r] += bv;
    }

    // online softmax (rows live in 16-lane halves; xor masks 1..8 stay in-half)
    float pr[2][8];
    #pragma unroll
    for (int r = 0; r < 8; ++r){
      float mx = fmaxf(sim[0][r], sim[1][r]);
      #pragma unroll
      for (int off = 8; off >= 1; off >>= 1) mx = fmaxf(mx, __shfl_xor(mx, off, 32));
      float mn = fmaxf(mrow[r], mx);
      float sc = __expf(mrow[r] - mn);
      mrow[r] = mn;
      float p0 = __expf(sim[0][r] - mn);
      float p1 = __expf(sim[1][r] - mn);
      pr[0][r] = p0; pr[1][r] = p1;
      float sum = p0 + p1;
      #pragma unroll
      for (int off = 8; off >= 1; off >>= 1) sum += __shfl_xor(sum, off, 32);
      lrow[r] = lrow[r]*sc + sum;
      #pragma unroll
      for (int t = 0; t < 4; ++t) acc[t][r] *= sc;
    }

    // P: C-layout (row=r+8*half, col=lane%16) -> LDS -> A-layout (row=lane%16, K=j)
    bf16* pp = &sP[wid][0];
    #pragma unroll
    for (int f = 0; f < 2; ++f)
      #pragma unroll
      for (int r = 0; r < 8; ++r)
        pp[(r + 8*half)*32 + 16*f + l16] = (bf16)pr[f][r];
    asm volatile("s_wait_dscnt 0x0" ::: "memory");
    v16bf pa = join8(*(const v8bf*)(pp + l16*32 + 16*half),
                     *(const v8bf*)(pp + l16*32 + 16*half + 8));
    #pragma unroll
    for (int t = 0; t < 4; ++t){
      v16bf vb = *(const v16bf*)&sV[buf][(16*t + l16)*32 + 16*half];
      acc[t] = wmma_bf16(pa, vb, acc[t]);
    }
    __syncthreads();                                          // buf may be overwritten next iter+1
  }

  #pragma unroll
  for (int r = 0; r < 8; ++r) lrow[r] = 1.f / lrow[r];
  #pragma unroll
  for (int t = 0; t < 4; ++t)
    #pragma unroll
    for (int r = 0; r < 8; ++r)
      o[((size_t)b*N_ + i0 + r + 8*half)*DIM_ + h*DH_ + 16*t + l16] = (bf16)(acc[t][r] * lrow[r]);
}

// ----- output projection: o(8192x512) @ WoutT-as-B -> f32 d_out (pre-LN) -----
__global__ __launch_bounds__(256) void k_out(const bf16* __restrict__ o, const bf16* __restrict__ WoutT,
                                             float* __restrict__ y){
  int gw = (int)((blockIdx.x * blockDim.x + threadIdx.x) >> 5);
  int lane = threadIdx.x & 31;
  int half = lane >> 4, l16 = lane & 15;
  int mt = gw >> 5, nt = gw & 31;
  int m0 = mt << 4, n0 = nt << 4;
  const bf16* arow = o + (size_t)(m0 + l16) * DIM_;
  const bf16* brow = WoutT + (size_t)(n0 + l16) * DIM_;
  v8f acc = {};
  #pragma unroll
  for (int w = 0; w < 16; ++w){
    v8bf a0 = *(const v8bf*)(arow + 32*w + 16*half);
    v8bf a1 = *(const v8bf*)(arow + 32*w + 16*half + 8);
    v16bf bv = *(const v16bf*)(brow + 32*w + 16*half);
    acc = wmma_bf16(join8(a0, a1), bv, acc);
  }
  #pragma unroll
  for (int r = 0; r < 8; ++r)
    y[(size_t)(m0 + r + 8*half)*DIM_ + n0 + l16] = acc[r];
}

extern "C" void kernel_launch(void* const* d_in, const int* in_sizes, int n_in,
                              void* d_out, int out_size, void* d_ws, size_t ws_size,
                              hipStream_t stream){
  (void)in_sizes; (void)n_in; (void)out_size; (void)ws_size;
  const float* x            = (const float*)d_in[0];
  const unsigned char* mask = (const unsigned char*)d_in[1];
  const float* gamma        = (const float*)d_in[2];
  const float* Wq           = (const float*)d_in[3];
  const float* Wkv          = (const float*)d_in[4];
  const float* nullkv       = (const float*)d_in[5];
  const float* Wout         = (const float*)d_in[6];
  const float* outg         = (const float*)d_in[7];
  float* y = (float*)d_out;

  char* ws = (char*)d_ws;
  size_t off = 0;
  bf16* xn    = (bf16*)(ws + off); off += (size_t)B_*N_*DIM_*2;      // 8 MB (reused as attn output)
  bf16* Wcat  = (bf16*)(ws + off); off += (size_t)640*512*2;
  bf16* WoT   = (bf16*)(ws + off); off += (size_t)512*512*2;
  bf16* qb    = (bf16*)(ws + off); off += (size_t)B_*H_*N_*DH_*2;    // 8 MB
  bf16* Kb    = (bf16*)(ws + off); off += (size_t)B_*JP*DH_*2;
  bf16* Vb    = (bf16*)(ws + off); off += (size_t)B_*DH_*JP*2;
  float* bias = (float*)(ws + off); off += (size_t)B_*JP*4;
  bf16* ob = xn;   // xn dead after k_proj; reuse for attention output

  k_ln1 <<<B_*N_, 128, 0, stream>>>(x, gamma, xn);
  k_wconv<<<(640*512 + 512*512)/256, 256, 0, stream>>>(Wq, Wkv, Wout, Wcat, WoT);
  k_prep<<<(B_*JP + 2*B_*DH_ + 2*B_*31*DH_ + 255)/256, 256, 0, stream>>>(mask, nullkv, bias, Kb, Vb);
  k_proj<<<(B_*N_/16)*40/8, 256, 0, stream>>>(xn, Wcat, qb, Kb, Vb);
  k_attn<<<B_*H_*(N_/64), 128, 0, stream>>>(qb, Kb, Vb, bias, ob);
  k_out <<<(B_*N_/16)*32/8, 256, 0, stream>>>(ob, WoT, y);
  k_ln2 <<<B_*N_, 128, 0, stream>>>(y, outg);
}
